// MultiHeadedAttention_88802743812238
// MI455X (gfx1250) — matrix operand order repaired and checked by reading, not compiled, not run on hardware
//
#include <hip/hip_runtime.h>
#include <hip/hip_bf16.h>
#include <math.h>

// ---------------------------------------------------------------------------
// MultiHeadedAttention (linear-attention variant) for MI455X / gfx1250.
// All matmuls use v_wmma_f32_16x16x32_f16 (wave32, 16x16 tiles, f32 accum).
// ---------------------------------------------------------------------------

typedef __attribute__((ext_vector_type(16))) _Float16 v16h;
typedef __attribute__((ext_vector_type(8)))  _Float16 v8h;
typedef __attribute__((ext_vector_type(8)))  float    v8f;
typedef _Float16 half_t;

#define N_TOK  4096
#define DMODEL 512
#define OUTD   256
#define NHEAD  4
#define DK     64

static __device__ __forceinline__ v8f wmma_f16(v16h a, v16h b, v8f c) {
  // (neg_a, A, neg_b, B, c_mod, C, reuse_a, reuse_b)
  return __builtin_amdgcn_wmma_f32_16x16x32_f16(false, a, false, b, (short)0, c,
                                                false, false);
}

// ---- Fragment loaders, contiguous k order K=16*(L/16)+j in both A and B (wave32) ----------
// A 16x32 (f16): lane L holds row M=L%16. Halves j -> K=(L/16)*16+j.
static __device__ __forceinline__ v16h
load_a_f32(const float* __restrict__ src, long ld, int m0, int k0, int lane) {
  const float* p0 = src + (long)(m0 + (lane & 15)) * ld + k0 + ((lane >> 4) << 4);
  v16h a;
#pragma unroll
  for (int j = 0; j < 8; ++j) {
    a[j]     = (half_t)p0[j];
    a[j + 8] = (half_t)p0[j + 8];
  }
  return a;
}

static __device__ __forceinline__ v16h
load_a_f16(const half_t* __restrict__ src, long ld, int m0, int k0, int lane) {
  const half_t* p0 = src + (long)(m0 + (lane & 15)) * ld + k0 + ((lane >> 4) << 4);
  v16h a;
#pragma unroll
  for (int j = 0; j < 8; ++j) {
    a[j]     = p0[j];
    a[j + 8] = p0[j + 8];
  }
  return a;
}

// A fragment sourced from a staged LDS tile [16 rows][32 K] (f16, row-major).
static __device__ __forceinline__ v16h
frag_a_lds(const half_t* tile, int lane) {
  const half_t* p0 = tile + (lane & 15) * 32 + ((lane >> 4) << 4);
  v16h a;
#pragma unroll
  for (int j = 0; j < 8; ++j) {
    a[j]     = p0[j];
    a[j + 8] = p0[j + 8];
  }
  return a;
}

// B 32x16 (f16): lane L holds column N=L%16; halves j -> K=(L/16)*16+j.
// K contiguous in memory rows (B(kk,n) = src[(n0+n)*ld + k0+kk]).
static __device__ __forceinline__ v16h
load_b_kcontig_f32(const float* __restrict__ src, long ld, int n0, int k0, int lane) {
  const float* p = src + (long)(n0 + (lane & 15)) * ld + k0 + ((lane >> 4) << 4);
  v16h b;
#pragma unroll
  for (int j = 0; j < 16; ++j) b[j] = (half_t)p[j];
  return b;
}

// Vectorized f16 variant: 32 contiguous bytes per lane (two b128 loads).
static __device__ __forceinline__ v16h
load_b_kcontig_f16(const half_t* __restrict__ src, long ld, int n0, int k0, int lane) {
  const half_t* p = src + (long)(n0 + (lane & 15)) * ld + k0 + ((lane >> 4) << 4);
  const v8h lo = *(const v8h*)p;
  const v8h hi = *(const v8h*)(p + 8);
  v16h b;
#pragma unroll
  for (int j = 0; j < 8; ++j) { b[j] = lo[j]; b[j + 8] = hi[j]; }
  return b;
}

// ---------------------------------------------------------------------------
// Kernel 1: projections  Y = X @ W^T + b
//   pidx 0/1 (q, k)  -> row-major f16 [N_TOK][OUTD]   (q: WMMA A, k: WMMA B later)
//   pidx 2/3 (qv,kv) -> transposed f16 [OUTD][N_TOK]  (WMMA B operands later)
// grid (OUTD/16, N_TOK/64, 4), block 128 (4 waves, 1 tile each)
// ---------------------------------------------------------------------------
__global__ __launch_bounds__(128) void proj_kernel(
    const float* __restrict__ query, const float* __restrict__ key,
    const float* __restrict__ Wq, const float* __restrict__ bq,
    const float* __restrict__ Wk, const float* __restrict__ bk,
    const float* __restrict__ Wqv, const float* __restrict__ bqv,
    const float* __restrict__ Wkv, const float* __restrict__ bkv,
    half_t* __restrict__ qh, half_t* __restrict__ kh,
    half_t* __restrict__ qvT, half_t* __restrict__ kvT) {
  const int pidx = blockIdx.z;
  const float* X; const float* W; const float* B;
  if (pidx == 0)      { X = query; W = Wq;  B = bq;  }
  else if (pidx == 1) { X = key;   W = Wk;  B = bk;  }
  else if (pidx == 2) { X = query; W = Wqv; B = bqv; }
  else                { X = key;   W = Wkv; B = bkv; }

  const int lane = threadIdx.x & 31;
  const int wave = threadIdx.x >> 5;
  const int n0 = (blockIdx.y * 4 + wave) * 16;
  const int o0 = blockIdx.x * 16;

  v8f acc = {};
  for (int kb = 0; kb < DMODEL; kb += 32) {
    v16h a = load_a_f32(X, DMODEL, n0, kb, lane);
    v16h b = load_b_kcontig_f32(W, DMODEL, o0, kb, lane);
    acc = wmma_f16(a, b, acc);
  }
  const int   col   = lane & 15;
  const int   mbase = (lane >> 4) * 8;
  const float bv    = B[o0 + col];

  if (pidx < 2) {
    half_t* D = (pidx == 0) ? qh : kh;
#pragma unroll
    for (int r = 0; r < 8; ++r)
      D[(long)(n0 + mbase + r) * OUTD + o0 + col] = (half_t)(acc[r] + bv);
  } else {
    // Transposed store: 8 consecutive n per lane -> one 16B packed store.
    half_t* DT = (pidx == 2) ? qvT : kvT;
    v8h pk;
#pragma unroll
    for (int r = 0; r < 8; ++r) pk[r] = (half_t)(acc[r] + bv);
    *(v8h*)&DT[(long)(o0 + col) * N_TOK + n0 + mbase] = pk;
  }
}

// ---------------------------------------------------------------------------
// Kernel 2: scores + masked softmax -> p[h][q][k] (f32, d_out region)
// One block = 16 q-rows of one head. Online softmax in registers, lane
// reduction via shfl, cross-wave via LDS, then one normalize pass over p.
// grid (N_TOK/16, NHEAD), block 256 (8 waves; each wave covers 32 col-tiles)
// ---------------------------------------------------------------------------
__global__ __launch_bounds__(256) void attn_softmax_kernel(
    const half_t* __restrict__ qh, const half_t* __restrict__ kh,
    const int* __restrict__ mask, float* __restrict__ p) {
  const int head = blockIdx.y;
  const int q0   = blockIdx.x * 16;
  const int lane = threadIdx.x & 31;
  const int wave = threadIdx.x >> 5;
  float* prow = p + (long)head * N_TOK * N_TOK;

  // q-row A fragments (d_k = 64 -> two K=32 fragments), shared by all tiles
  const v16h a0 = load_a_f16(qh, OUTD, q0, head * DK + 0, lane);
  const v16h a1 = load_a_f16(qh, OUTD, q0, head * DK + 32, lane);

  const int col   = lane & 15;
  const int mbase = (lane >> 4) * 8;

  float mrun[8], lrun[8];
#pragma unroll
  for (int r = 0; r < 8; ++r) { mrun[r] = -INFINITY; lrun[r] = 0.f; }

  // Pass 1: scores, mask, raw store + online (m,l)
  for (int t = wave; t < N_TOK / 16; t += 8) {
    const int k0 = t * 16;
    // Prefetch next k-tile row for this lane (gfx1250 global_prefetch_b8).
    __builtin_prefetch(kh + (long)(k0 + 128 + col) * OUTD + head * DK, 0, 0);

    v8f acc = {};
    v16h b0 = load_b_kcontig_f16(kh, OUTD, k0, head * DK + 0, lane);
    acc = wmma_f16(a0, b0, acc);
    v16h b1 = load_b_kcontig_f16(kh, OUTD, k0, head * DK + 32, lane);
    acc = wmma_f16(a1, b1, acc);

    const int ki = k0 + col;
#pragma unroll
    for (int r = 0; r < 8; ++r) {
      const int qrow = q0 + mbase + r;
      float s = acc[r] * 0.125f;  // 1/sqrt(64)
      if (mask[(long)qrow * N_TOK + ki] == 0) s = -1e9f;
      prow[(long)qrow * N_TOK + ki] = s;
      const float mn = fmaxf(mrun[r], s);
      lrun[r] = lrun[r] * __expf(mrun[r] - mn) + __expf(s - mn);
      mrun[r] = mn;
    }
  }

  // Reduce (m,l) across the 16 lanes of each half-wave (same rows, disjoint cols)
#pragma unroll
  for (int r = 0; r < 8; ++r) {
    float m = mrun[r], l = lrun[r];
#pragma unroll
    for (int off = 1; off < 16; off <<= 1) {
      const float mo = __shfl_xor(m, off, 32);
      const float lo = __shfl_xor(l, off, 32);
      const float mn = fmaxf(m, mo);
      l = l * __expf(m - mn) + lo * __expf(mo - mn);
      m = mn;
    }
    mrun[r] = m; lrun[r] = l;
  }

  __shared__ float lm[8][16], ll[8][16], fm[16], fl[16];
  if (col == 0) {
#pragma unroll
    for (int r = 0; r < 8; ++r) {
      lm[wave][mbase + r] = mrun[r];
      ll[wave][mbase + r] = lrun[r];
    }
  }
  __threadfence();   // make pass-1 p stores visible to whole block
  __syncthreads();

  if (threadIdx.x < 16) {
    const int row = threadIdx.x;
    float m = lm[0][row], l = ll[0][row];
#pragma unroll
    for (int w = 1; w < 8; ++w) {
      const float mo = lm[w][row], lo = ll[w][row];
      const float mn = fmaxf(m, mo);
      l = l * __expf(m - mn) + lo * __expf(mo - mn);
      m = mn;
    }
    fm[row] = m;
    fl[row] = 1.0f / l;
  }
  __syncthreads();

  // Pass 2: coalesced normalize of the 16 x 4096 slab
  float* slab = prow + (long)q0 * N_TOK;
  for (int idx = threadIdx.x; idx < 16 * N_TOK; idx += 256) {
    const int row = idx >> 12;          // N_TOK == 4096 == 2^12
    const int c   = idx & (N_TOK - 1);
    const float v = slab[(long)row * N_TOK + c];
    slab[(long)row * N_TOK + c] = __expf(v - fm[row]) * fl[row];
  }
}

// ---------------------------------------------------------------------------
// Kernel 3: x1 = ELU(p @ kv)    out[n][h*64+d]
// Block stages the shared 16x32 p-tile into LDS once (coalesced, f16);
// all 4 waves (4 d-tiles) build their A fragment from LDS.
// grid (N_TOK/16, NHEAD), block 128
// ---------------------------------------------------------------------------
__global__ __launch_bounds__(128) void x1_kernel(
    const float* __restrict__ p, const half_t* __restrict__ kvT,
    float* __restrict__ x1) {
  const int head = blockIdx.y;
  const int q0   = blockIdx.x * 16;
  const int lane = threadIdx.x & 31;
  const int wave = threadIdx.x >> 5;
  const int o0   = head * DK + wave * 16;
  const float* prow = p + (long)head * N_TOK * N_TOK;

  __shared__ half_t atile[16 * 32];

  v8f acc = {};
  for (int kb = 0; kb < N_TOK; kb += 32) {
    __syncthreads();
    // Coalesced cooperative stage: p[q0..q0+15][kb..kb+31] -> LDS f16
    for (int i = threadIdx.x; i < 512; i += 128) {
      const int kk = i & 31, m = i >> 5;
      atile[m * 32 + kk] = (half_t)prow[(long)(q0 + m) * N_TOK + kb + kk];
      __builtin_prefetch(prow + (long)(q0 + m) * N_TOK + kb + 64 + kk, 0, 0);
    }
    __syncthreads();

    v16h a = frag_a_lds(atile, lane);
    v16h b = load_b_kcontig_f16(kvT, N_TOK, o0, kb, lane);
    acc = wmma_f16(a, b, acc);
  }
  const int col = lane & 15, mbase = (lane >> 4) * 8;
#pragma unroll
  for (int r = 0; r < 8; ++r) {
    float v = acc[r];
    v = v > 0.f ? v : (__expf(v) - 1.f);   // ELU
    x1[(long)(q0 + mbase + r) * OUTD + o0 + col] = v;
  }
}

// ---------------------------------------------------------------------------
// Kernel 4: x2 = ELU(p^T @ qv)  out[k][h*64+d]
// The p^T tile is staged through LDS with a coalesced read (16 consecutive
// floats per q-row), so the transpose costs no strided global traffic.
// grid (N_TOK/16, NHEAD), block 128
// ---------------------------------------------------------------------------
__global__ __launch_bounds__(128) void x2_kernel(
    const float* __restrict__ p, const half_t* __restrict__ qvT,
    float* __restrict__ x2) {
  const int head = blockIdx.y;
  const int k0   = blockIdx.x * 16;
  const int lane = threadIdx.x & 31;
  const int wave = threadIdx.x >> 5;
  const int o0   = head * DK + wave * 16;
  const float* prow = p + (long)head * N_TOK * N_TOK;

  __shared__ half_t atile[16 * 32];   // [m = k-col][kk = q-row]

  v8f acc = {};
  for (int qb = 0; qb < N_TOK; qb += 32) {
    __syncthreads();
    // Coalesced stage of transposed tile: p[qb+kk][k0+m] -> atile[m][kk]
    for (int i = threadIdx.x; i < 512; i += 128) {
      const int m = i & 15, kk = i >> 4;
      atile[m * 32 + kk] = (half_t)prow[(long)(qb + kk) * N_TOK + k0 + m];
      __builtin_prefetch(prow + (long)(qb + 32 + kk) * N_TOK + k0 + m, 0, 0);
    }
    __syncthreads();

    v16h a = frag_a_lds(atile, lane);
    v16h b = load_b_kcontig_f16(qvT, N_TOK, o0, qb, lane);
    acc = wmma_f16(a, b, acc);
  }
  const int col = lane & 15, mbase = (lane >> 4) * 8;
#pragma unroll
  for (int r = 0; r < 8; ++r) {
    float v = acc[r];
    v = v > 0.f ? v : (__expf(v) - 1.f);   // ELU
    x2[(long)(k0 + mbase + r) * OUTD + o0 + col] = v;
  }
}

// ---------------------------------------------------------------------------
extern "C" void kernel_launch(void* const* d_in, const int* in_sizes, int n_in,
                              void* d_out, int out_size, void* d_ws, size_t ws_size,
                              hipStream_t stream) {
  (void)in_sizes; (void)n_in; (void)out_size; (void)ws_size;
  const float* query = (const float*)d_in[0];
  const float* key   = (const float*)d_in[1];
  const int*   mask  = (const int*)d_in[2];
  const float* Wq  = (const float*)d_in[3];
  const float* bq  = (const float*)d_in[4];
  const float* Wk  = (const float*)d_in[5];
  const float* bk  = (const float*)d_in[6];
  const float* Wqv = (const float*)d_in[7];
  const float* bqv = (const float*)d_in[8];
  const float* Wkv = (const float*)d_in[9];
  const float* bkv = (const float*)d_in[10];

  float* x1 = (float*)d_out;                       // [N_TOK, OUTD]
  float* x2 = x1 + (long)N_TOK * OUTD;             // [N_TOK, OUTD]
  float* p  = x2 + (long)N_TOK * OUTD;             // [NHEAD, N_TOK, N_TOK]

  // f16 workspace: qh, kh row-major; qvT, kvT transposed. 4 x 2 MB = 8 MB.
  half_t* qh  = (half_t*)d_ws;
  half_t* kh  = qh  + (long)N_TOK * OUTD;
  half_t* qvT = kh  + (long)N_TOK * OUTD;
  half_t* kvT = qvT + (long)N_TOK * OUTD;

  proj_kernel<<<dim3(OUTD / 16, N_TOK / 64, 4), 128, 0, stream>>>(
      query, key, Wq, bq, Wk, bk, Wqv, bqv, Wkv, bkv, qh, kh, qvT, kvT);

  attn_softmax_kernel<<<dim3(N_TOK / 16, NHEAD), 256, 0, stream>>>(qh, kh, mask, p);

  x1_kernel<<<dim3(N_TOK / 16, NHEAD), 128, 0, stream>>>(p, kvT, x1);
  x2_kernel<<<dim3(N_TOK / 16, NHEAD), 128, 0, stream>>>(p, qvT, x2);
}
